// SAModule_609885356786
// MI455X (gfx1250) — matrix operand order repaired and checked by reading, not compiled, not run on hardware
//
#include <hip/hip_runtime.h>
#include <hip/hip_bf16.h>

#define BATCH 8
#define NPTS  4096
#define NSAMP 1024          // S
#define CIN   64
#define HIDN  128
#define K1PAD 96            // 67 padded up to 3 K-tiles of 32
#define MAXNB 64
#define NCENT (BATCH * NSAMP)   // 8192

typedef _Float16 v16h __attribute__((ext_vector_type(16)));
typedef _Float16 v8h  __attribute__((ext_vector_type(8)));
typedef float    v8f  __attribute__((ext_vector_type(8)));

union F16Frag { v16h v; v8h h[2]; };

// ---------------------------------------------------------------------------
// Kernel 0: transpose + f16-convert weights.  W1 [67,128] -> W1T f16 [128][96]
// (zero-padded K), W2 [128,128] -> W2T f16 [128][128].
// ---------------------------------------------------------------------------
__global__ __launch_bounds__(128) void prep_weights(const float* __restrict__ W1,
                                                    const float* __restrict__ W2,
                                                    _Float16* __restrict__ W1T,
                                                    _Float16* __restrict__ W2T) {
    int n = blockIdx.x;                 // output column 0..127
    for (int k = threadIdx.x; k < K1PAD; k += blockDim.x)
        W1T[n * K1PAD + k] = (k < CIN + 3) ? (_Float16)W1[k * HIDN + n] : (_Float16)0.0f;
    for (int k = threadIdx.x; k < HIDN; k += blockDim.x)
        W2T[n * HIDN + k] = (_Float16)W2[k * HIDN + n];
}

// ---------------------------------------------------------------------------
// Kernel 1: farthest point sampling.  One block per cloud; each of 256 threads
// owns 16 points with register-resident min-d2; argmax via LDS + wave reduce.
// Writes centers (pos of selected idx) straight into the output tensor.
// ---------------------------------------------------------------------------
__global__ __launch_bounds__(256) void fps_kernel(const float* __restrict__ pos,
                                                  float* __restrict__ cent_out) {
    const int b   = blockIdx.x;
    const int tid = threadIdx.x;
    const float* p = pos + (size_t)b * NPTS * 3;

    float px[16], py[16], pz[16], mind[16];
#pragma unroll
    for (int j = 0; j < 16; ++j) {
        int i = tid + 256 * j;
        px[j] = p[i * 3 + 0];
        py[j] = p[i * 3 + 1];
        pz[j] = p[i * 3 + 2];
        mind[j] = 1e10f;
    }

    __shared__ float s_lx, s_ly, s_lz;
    __shared__ float s_bd[256];
    __shared__ int   s_bi[256];

    if (tid == 0) {
        s_lx = p[0]; s_ly = p[1]; s_lz = p[2];
        float* c0 = cent_out + (size_t)b * NSAMP * 3;
        c0[0] = p[0]; c0[1] = p[1]; c0[2] = p[2];   // idx[0] = 0
    }
    __syncthreads();

    for (int s = 1; s < NSAMP; ++s) {
        const float lx = s_lx, ly = s_ly, lz = s_lz;
        float best = -1.0f; int bi = 0;
#pragma unroll
        for (int j = 0; j < 16; ++j) {
            float dx = px[j] - lx, dy = py[j] - ly, dz = pz[j] - lz;
            float d2 = dx * dx + dy * dy + dz * dz;
            float m  = fminf(mind[j], d2);
            mind[j]  = m;
            if (m > best) { best = m; bi = tid + 256 * j; }
        }
        s_bd[tid] = best; s_bi[tid] = bi;
        __syncthreads();
        if (tid < 32) {
            float bb = s_bd[tid]; int ii = s_bi[tid];
            for (int k = tid + 32; k < 256; k += 32) {
                float ob = s_bd[k]; int oi = s_bi[k];
                if (ob > bb || (ob == bb && oi < ii)) { bb = ob; ii = oi; }
            }
#pragma unroll
            for (int off = 16; off > 0; off >>= 1) {
                float ob = __shfl_down(bb, off);
                int   oi = __shfl_down(ii, off);
                if (ob > bb || (ob == bb && oi < ii)) { bb = ob; ii = oi; }
            }
            if (tid == 0) {
                float nx = p[ii * 3 + 0], ny = p[ii * 3 + 1], nz = p[ii * 3 + 2];
                s_lx = nx; s_ly = ny; s_lz = nz;
                float* co = cent_out + ((size_t)b * NSAMP + s) * 3;
                co[0] = nx; co[1] = ny; co[2] = nz;
            }
        }
        __syncthreads();
    }
}

// ---------------------------------------------------------------------------
// Kernel 2: radius ball query.  One wave (32 lanes) per center; ordered append
// of the first MAXNB in-range point indices via ballot + prefix popcount
// (matches reference's stable argsort-of-~valid semantics).  Also emits
// batch_out.
// ---------------------------------------------------------------------------
__global__ __launch_bounds__(256) void ballq_kernel(const float* __restrict__ pos,
                                                    const float* __restrict__ cent,
                                                    int* __restrict__ nb,
                                                    int* __restrict__ ncnt,
                                                    float* __restrict__ batch_out) {
    const int wave   = threadIdx.x >> 5;
    const int lane   = threadIdx.x & 31;
    const int center = blockIdx.x * 8 + wave;
    const int b      = center >> 10;                 // / NSAMP
    const float r2   = 0.2f * 0.2f;

    const float cx = cent[center * 3 + 0];
    const float cy = cent[center * 3 + 1];
    const float cz = cent[center * 3 + 2];
    const float* p = pos + (size_t)b * NPTS * 3;

    int count = 0;
    for (int base = 0; base < NPTS && count < MAXNB; base += 32) {
        const int i = base + lane;
        float dx = p[i * 3 + 0] - cx;
        float dy = p[i * 3 + 1] - cy;
        float dz = p[i * 3 + 2] - cz;
        bool valid = (dx * dx + dy * dy + dz * dz) <= r2;
        unsigned bal = __builtin_amdgcn_ballot_w32(valid);
        int pre  = __popc(bal & ((1u << lane) - 1u));
        int slot = count + pre;
        if (valid && slot < MAXNB) nb[(size_t)center * MAXNB + slot] = i;
        count += __popc(bal);
    }
    if (count > MAXNB) count = MAXNB;
    for (int s = count + lane; s < MAXNB; s += 32)   // pad with idx 0 (masked later)
        nb[(size_t)center * MAXNB + s] = 0;
    if (lane == 0) {
        ncnt[center]      = count;
        batch_out[center] = (float)b;
    }
}

// ---------------------------------------------------------------------------
// Kernel 3: PointConv MLP via WMMA.  One block (8 waves) per center.
// GEMM1: [64 x 96(pad67)] @ [96 x 128] ; GEMM2: [64 x 128] @ [128 x 128]
// f16 inputs, f32 accumulate, ReLU, masked max over neighbors.
// Wave w owns N-tile w (16 output columns); 4 M-tiles cover the 64 rows.
// ---------------------------------------------------------------------------
__global__ __launch_bounds__(256) void pointconv_kernel(const float* __restrict__ x,
                                                        const float* __restrict__ pos,
                                                        const float* __restrict__ cent,
                                                        const int* __restrict__ nb,
                                                        const int* __restrict__ ncnt,
                                                        const _Float16* __restrict__ W1T,
                                                        const _Float16* __restrict__ W2T,
                                                        const float* __restrict__ b1,
                                                        const float* __restrict__ b2,
                                                        float* __restrict__ out) {
    __shared__ __align__(16) _Float16 sA[MAXNB * K1PAD];   // 12 KB gathered inputs
    __shared__ __align__(16) _Float16 sH[MAXNB * HIDN];    // 16 KB hidden acts
    __shared__ int sNb[MAXNB];

    const int center = blockIdx.x;
    const int b      = center >> 10;
    const int tid    = threadIdx.x;
    const int wave   = tid >> 5;        // N-tile
    const int lane   = tid & 31;
    const int half   = lane >> 4;       // 0 / 1 (K- or M-half per ISA layout)
    const int mrow   = lane & 15;

    if (tid < MAXNB) sNb[tid] = nb[(size_t)center * MAXNB + tid];
    __syncthreads();

    const float cx = cent[center * 3 + 0];
    const float cy = cent[center * 3 + 1];
    const float cz = cent[center * 3 + 2];
    const float* xb = x   + (size_t)b * NPTS * CIN;
    const float* pb = pos + (size_t)b * NPTS * 3;

    // ---- gather: 4 threads per neighbor row, 24 features each ----
    {
        const int row  = tid >> 2;
        const int part = tid & 3;
        const int j    = sNb[row];
        const float* xr = xb + (size_t)j * CIN;
        for (int c = part * 24; c < part * 24 + 24; ++c) {
            float v;
            if (c < CIN)            v = xr[c];
            else if (c == CIN)      v = pb[j * 3 + 0] - cx;
            else if (c == CIN + 1)  v = pb[j * 3 + 1] - cy;
            else if (c == CIN + 2)  v = pb[j * 3 + 2] - cz;
            else                    v = 0.0f;
            sA[row * K1PAD + c] = (_Float16)v;
        }
    }
    __syncthreads();

    // ---- GEMM1: sA [64 x 96] @ W1T-as-B [96 x 128] ----
    v8f c1[4] = {};
#pragma unroll
    for (int kt = 0; kt < 3; ++kt) {
        F16Frag Bf;
        const _Float16* wrow = W1T + (size_t)(wave * 16 + mrow) * K1PAD + kt * 32 + 8 * half;
        Bf.h[0] = ((const v8h*)wrow)[0];
        Bf.h[1] = ((const v8h*)wrow)[2];
#pragma unroll
        for (int t = 0; t < 4; ++t) {
            F16Frag Af;
            const _Float16* arow = sA + (t * 16 + mrow) * K1PAD + kt * 32 + 8 * half;
            Af.h[0] = *(const v8h*)(arow);
            Af.h[1] = *(const v8h*)(arow + 16);
            c1[t] = __builtin_amdgcn_wmma_f32_16x16x32_f16(
                false, Af.v, false, Bf.v, (short)0, c1[t], false, false);
        }
    }

    // bias + ReLU, stage hidden activations to LDS (f16, row-major [M][128])
    {
        const float bias1 = b1[wave * 16 + mrow];
#pragma unroll
        for (int t = 0; t < 4; ++t) {
#pragma unroll
            for (int r = 0; r < 8; ++r) {
                float v = fmaxf(c1[t][r] + bias1, 0.0f);
                int M = t * 16 + half * 8 + r;
                sH[M * HIDN + wave * 16 + mrow] = (_Float16)v;
            }
        }
    }
    __syncthreads();

    // ---- GEMM2: sH [64 x 128] @ W2T-as-B [128 x 128] ----
    v8f c2[4] = {};
#pragma unroll
    for (int kt = 0; kt < 4; ++kt) {
        F16Frag Bf;
        const _Float16* wrow = W2T + (size_t)(wave * 16 + mrow) * HIDN + kt * 32 + 8 * half;
        Bf.h[0] = ((const v8h*)wrow)[0];
        Bf.h[1] = ((const v8h*)wrow)[2];
#pragma unroll
        for (int t = 0; t < 4; ++t) {
            F16Frag Af;
            const _Float16* arow = sH + (t * 16 + mrow) * HIDN + kt * 32 + 8 * half;
            Af.h[0] = *(const v8h*)(arow);
            Af.h[1] = *(const v8h*)(arow + 16);
            c2[t] = __builtin_amdgcn_wmma_f32_16x16x32_f16(
                false, Af.v, false, Bf.v, (short)0, c2[t], false, false);
        }
    }

    // ---- bias + ReLU + validity mask + max over the 64 neighbor rows ----
    const float bias2 = b2[wave * 16 + mrow];
    const int   nc    = ncnt[center];
    float vmax = -__builtin_inff();
#pragma unroll
    for (int t = 0; t < 4; ++t) {
#pragma unroll
        for (int r = 0; r < 8; ++r) {
            int M = t * 16 + half * 8 + r;
            float v = fmaxf(c2[t][r] + bias2, 0.0f);
            if (M < nc && v > vmax) vmax = v;
        }
    }
    float other = __shfl_xor(vmax, 16);          // merge M-halves (same N column)
    vmax = fmaxf(vmax, other);
    if (half == 0)
        out[(size_t)center * HIDN + wave * 16 + mrow] = vmax;
}

// ---------------------------------------------------------------------------
extern "C" void kernel_launch(void* const* d_in, const int* in_sizes, int n_in,
                              void* d_out, int out_size, void* d_ws, size_t ws_size,
                              hipStream_t stream) {
    const float* x   = (const float*)d_in[0];   // [B*N, 64]
    const float* pos = (const float*)d_in[1];   // [B*N, 3]
    // d_in[2] = batch (unused; implied by layout)
    const float* W1  = (const float*)d_in[3];   // [67, 128]
    const float* b1  = (const float*)d_in[4];   // [128]
    const float* W2  = (const float*)d_in[5];   // [128, 128]
    const float* b2  = (const float*)d_in[6];   // [128]

    // output partitions: x_out [8192,128] | cent [8192,3] | batch [8192]
    float* out_x     = (float*)d_out;
    float* out_cent  = out_x + (size_t)NCENT * HIDN;
    float* out_batch = out_cent + (size_t)NCENT * 3;

    // workspace partitions (16B aligned)
    char* ws = (char*)d_ws;
    int*      nb   = (int*)(ws);                                  // 8192*64*4 = 2 MB
    int*      ncnt = (int*)(ws + (size_t)NCENT * MAXNB * 4);      // 32 KB
    _Float16* W1T  = (_Float16*)(ws + (size_t)NCENT * MAXNB * 4 + (size_t)NCENT * 4);
    _Float16* W2T  = W1T + (size_t)HIDN * K1PAD;

    prep_weights<<<HIDN, 128, 0, stream>>>(W1, W2, W1T, W2T);
    fps_kernel<<<BATCH, 256, 0, stream>>>(pos, out_cent);
    ballq_kernel<<<NCENT / 8, 256, 0, stream>>>(pos, out_cent, nb, ncnt, out_batch);
    pointconv_kernel<<<NCENT, 256, 0, stream>>>(x, pos, out_cent, nb, ncnt,
                                                W1T, W2T, b1, b2, out_x);
}
